// HyperGraph_68942815035527
// MI455X (gfx1250) — matrix operand order repaired and checked by reading, not compiled, not run on hardware
//
#include <hip/hip_runtime.h>

// ---------------------------------------------------------------------------
// CDNA5 (gfx1250) wave32 WMMA implementation of the HyperGraph layer.
// All GEMMs use v_wmma_f32_16x16x32_bf16, 4 N-tiles register-blocked per wave;
// weights pre-packed to bf16 in WMMA B-fragment lane order (pure b128 loads).
// ---------------------------------------------------------------------------

typedef __bf16 bf16_t;
typedef __attribute__((ext_vector_type(16))) __bf16 v16bf;
typedef __attribute__((ext_vector_type(8)))  float  v8f;

static __device__ __forceinline__ v8f wmma_bf16(v16bf a, v16bf b, v8f c) {
  // 8 args: (neg_a, A, neg_b, B, c_mod, C, reuse_a, reuse_b)
  return __builtin_amdgcn_wmma_f32_16x16x32_bf16(false, a, false, b, (short)0, c,
                                                 false, false);
}

// ---- fragment loaders (ISA 7.12.2 16-bit layouts, wave32) ------------------
// A (16x32, MxK): lane L -> M = L&15 ; half=L>>4
//   VGPR v<4 : K = 2v   + 8*half + {0,1}   -> contiguous run [8*half, 8*half+8)
//   VGPR v>=4: K = 16 + 2(v-4) + 8*half    -> contiguous run [16+8*half, +8)
static __device__ __forceinline__ v16bf
pack_a_from_runs(float4 q0, float4 q1, float4 q2, float4 q3) {
  v16bf a;
  a[0]  = (bf16_t)q0.x; a[1]  = (bf16_t)q0.y; a[2]  = (bf16_t)q0.z; a[3]  = (bf16_t)q0.w;
  a[4]  = (bf16_t)q1.x; a[5]  = (bf16_t)q1.y; a[6]  = (bf16_t)q1.z; a[7]  = (bf16_t)q1.w;
  a[8]  = (bf16_t)q2.x; a[9]  = (bf16_t)q2.y; a[10] = (bf16_t)q2.z; a[11] = (bf16_t)q2.w;
  a[12] = (bf16_t)q3.x; a[13] = (bf16_t)q3.y; a[14] = (bf16_t)q3.z; a[15] = (bf16_t)q3.w;
  return a;
}

static __device__ __forceinline__ v16bf
load_a_frag(const float* __restrict__ A, int lda, int m0, int kk, int lane) {
  const int m = m0 + (lane & 15);
  const int half = lane >> 4;
  const float* base = A + (size_t)m * lda + kk + 8 * half;   // 16B aligned
  const float4 q0 = ((const float4*)base)[0];
  const float4 q1 = ((const float4*)base)[1];
  const float4 q2 = ((const float4*)(base + 16))[0];
  const float4 q3 = ((const float4*)(base + 16))[1];
  return pack_a_from_runs(q0, q1, q2, q3);
}

// Transposed-A variant (A[m,k] = A[k*lda+m]) -- only used for tiny H^T GEMM.
static __device__ __forceinline__ v16bf
load_a_frag_t(const float* __restrict__ A, int lda, int m0, int kk, int lane) {
  v16bf a;
  const int m = m0 + (lane & 15);
  const int half = lane >> 4;
#pragma unroll
  for (int v = 0; v < 8; ++v) {
    const int k = kk + ((v < 4) ? (2 * v + 8 * half) : (16 + 2 * (v - 4) + 8 * half));
    a[2 * v]     = (bf16_t)A[(size_t)k * lda + m];
    a[2 * v + 1] = (bf16_t)A[(size_t)(k + 1) * lda + m];
  }
  return a;
}

// B (32x16, KxN) from f32 row-major: lane L -> N=L&15; half=L>>4;
// VGPR v: K = 2v + 8*half + (v>=4 ? 8 : 0) + {0,1}  (same as A)
static __device__ __forceinline__ v16bf
load_b_frag(const float* __restrict__ Bw, int ldb, int n0, int kk, int lane) {
  v16bf b;
  const int n = n0 + (lane & 15);
  const int half = lane >> 4;
  const float* base = Bw + (size_t)(kk + 8 * half) * ldb + n;
#pragma unroll
  for (int v = 0; v < 8; ++v) {
    b[2 * v]     = (bf16_t)base[(size_t)(2 * v + ((v >= 4) ? 8 : 0)) * ldb];
    b[2 * v + 1] = (bf16_t)base[(size_t)(2 * v + 1 + ((v >= 4) ? 8 : 0)) * ldb];
  }
  return b;
}

// B from pre-packed bf16 fragment store: [(kt*nt + ntile)*32 + lane][16]
static __device__ __forceinline__ v16bf
load_b_packed(const bf16_t* __restrict__ pB, int nt, int kt, int ntile, int lane) {
  return *(const v16bf*)(pB + (((size_t)kt * nt + ntile) * 32 + lane) * 16);
}

// D (16x16 f32): VGPR r, lane L -> row = r + 8*(L>>4), col = L&15
static __device__ __forceinline__ void
store_d(float* __restrict__ D, int ldd, int m0, int n0, int lane, v8f c,
        const float* __restrict__ bias, float scale, int relu) {
  const int n = n0 + (lane & 15);
  const int half = lane >> 4;
  const float bv = bias ? bias[n] : 0.0f;
#pragma unroll
  for (int r = 0; r < 8; ++r) {
    float v = c[r] * scale + bv;
    if (relu) v = fmaxf(v, 0.0f);
    D[(size_t)(m0 + r + 8 * half) * ldd + n] = v;
  }
}

// ---- weight packer: f32 [K,N] row-major -> bf16 B-fragment order -----------
__global__ void __launch_bounds__(256)
pack_w_kernel(const float* __restrict__ W, bf16_t* __restrict__ out,
              int N, int ntiles, int total) {
  const int idx = (int)(blockIdx.x * blockDim.x + threadIdx.x);
  if (idx >= total) return;
  const int i = idx & 15;
  const int lane = (idx >> 4) & 31;
  const int tile = idx >> 9;                 // kt*ntiles + ntile
  const int kt = tile / ntiles;
  const int ntile = tile - kt * ntiles;
  const int k = kt * 32 + (i & 7) + 8 * (lane >> 4) + 16 * (i >> 3);
  const int n = ntile * 16 + (lane & 15);
  out[idx] = (bf16_t)W[(size_t)k * N + n];
}

// ---- generic (batched) GEMM: D = act(scale * op(A)@B + bias) ---------------
// Each wave computes 16 x (16*NBLK) of D: one A fragment amortized over NBLK
// WMMAs per k-step.
template <int TRANSA, int PACKEDB, int NBLK>
__global__ void __launch_bounds__(256)
gemm_bf16_wmma(const float* __restrict__ A, const float* __restrict__ Bw,
               const bf16_t* __restrict__ pB,
               const float* __restrict__ bias, float* __restrict__ D,
               int M, int N, int K, int lda, int ldb, int ldd,
               long long sA, long long sB, long long sD, int batches,
               int relu, float scale) {
  const int lane = threadIdx.x & 31;
  const int wave = (int)((blockIdx.x * blockDim.x + threadIdx.x) >> 5);
  const int nt = N >> 4;
  const int ng = nt / NBLK;                     // N groups per row of tiles
  const int tilesPerBatch = (M >> 4) * ng;
  if (wave >= tilesPerBatch * batches) return;  // wave-uniform: EXEC all-1s
  const int bidx = wave / tilesPerBatch;
  const int t = wave - bidx * tilesPerBatch;
  const int m0 = (t / ng) * 16;
  const int ntile0 = (t % ng) * NBLK;
  const float* Ab = A + (size_t)bidx * sA;
  const float* Bb = Bw + (size_t)bidx * sB;
  float* Db = D + (size_t)bidx * sD;

  v8f acc[NBLK];
#pragma unroll
  for (int j = 0; j < NBLK; ++j) acc[j] = (v8f){0.f,0.f,0.f,0.f,0.f,0.f,0.f,0.f};

  const int kts = K >> 5;
  for (int kt = 0; kt < kts; ++kt) {
    v16bf a = TRANSA ? load_a_frag_t(Ab, lda, m0, kt * 32, lane)
                     : load_a_frag(Ab, lda, m0, kt * 32, lane);
    if (PACKEDB) {
      // next k-step's packed-B block is contiguous: one speculative prefetch
      __builtin_prefetch(pB + (((size_t)(kt + 1) * nt + ntile0) * 32 + lane) * 16, 0, 0);
    }
#pragma unroll
    for (int j = 0; j < NBLK; ++j) {
      v16bf b = PACKEDB ? load_b_packed(pB, nt, kt, ntile0 + j, lane)
                        : load_b_frag(Bb, ldb, (ntile0 + j) * 16, kt * 32, lane);
      acc[j] = wmma_bf16(a, b, acc[j]);
    }
  }
#pragma unroll
  for (int j = 0; j < NBLK; ++j)
    store_d(Db, ldd, m0, (ntile0 + j) * 16, lane, acc[j], bias, scale, relu);
}

// ---- fused attention MLP: aw = relu(relu([xm_n||ei_e]@Wa1+ba1)@Wa2+ba2) ----
// Row r = b*4096 + e*64 + n ; A[r,k] = k<256 ? xm[b,n,k] : edge_init[b,e,k-256]
__global__ void __launch_bounds__(256)
attn_mlp_wmma(const float* __restrict__ xm, const float* __restrict__ ei,
              const bf16_t* __restrict__ pWa1, const float* __restrict__ ba1,
              const float* __restrict__ Wa2, const float* __restrict__ ba2,
              float* __restrict__ aw, int totalWaves) {
  const int lane = threadIdx.x & 31;
  const int wave = (int)((blockIdx.x * blockDim.x + threadIdx.x) >> 5);
  if (wave >= totalWaves) return;
  const int r0 = wave * 16;
  const int half = lane >> 4;
  const int mloc = lane & 15;

  const int r = r0 + mloc;
  const int b = r >> 12, e = (r >> 6) & 63, n = r & 63;
  const float* xrow = xm + (size_t)(b * 64 + n) * 256;
  const float* erow = ei + (size_t)(b * 64 + e) * 256;

  v8f acc0 = {0.f, 0.f, 0.f, 0.f, 0.f, 0.f, 0.f, 0.f};
  v8f acc1 = {0.f, 0.f, 0.f, 0.f, 0.f, 0.f, 0.f, 0.f};
#pragma unroll 4
  for (int kt = 0; kt < 16; ++kt) {
    const int kk = kt * 32;
    // each 8-float run stays inside one concat region (256 % 8 == 0)
    const int kb0 = kk + 8 * half;
    const int kb1 = kk + 16 + 8 * half;
    const float* p0 = (kb0 < 256) ? (xrow + kb0) : (erow + kb0 - 256);
    const float* p1 = (kb1 < 256) ? (xrow + kb1) : (erow + kb1 - 256);
    const float4 q0 = ((const float4*)p0)[0];
    const float4 q1 = ((const float4*)p0)[1];
    const float4 q2 = ((const float4*)p1)[0];
    const float4 q3 = ((const float4*)p1)[1];
    v16bf a = pack_a_from_runs(q0, q1, q2, q3);

    v16bf b0 = load_b_packed(pWa1, 2, kt, 0, lane);
    v16bf b1 = load_b_packed(pWa1, 2, kt, 1, lane);
    acc0 = wmma_bf16(a, b0, acc0);
    acc1 = wmma_bf16(a, b1, acc1);
  }

  // fused second layer: relu(+ba1) * Wa2[col], then row-sum over 32 columns
  const float w0 = Wa2[mloc],      w1 = Wa2[16 + mloc];
  const float c0 = ba1[mloc],      c1 = ba1[16 + mloc];
  const float b2 = ba2[0];
  float s[8];
#pragma unroll
  for (int rr = 0; rr < 8; ++rr)
    s[rr] = fmaxf(acc0[rr] + c0, 0.f) * w0 + fmaxf(acc1[rr] + c1, 0.f) * w1;
#pragma unroll
  for (int sh = 1; sh < 16; sh <<= 1) {
#pragma unroll
    for (int rr = 0; rr < 8; ++rr) s[rr] += __shfl_xor(s[rr], sh, 32);
  }
  if (mloc == 0) {
#pragma unroll
    for (int rr = 0; rr < 8; ++rr)
      aw[(size_t)(r0 + rr + 8 * half)] = fmaxf(s[rr] + b2, 0.f);
  }
}

// ---- sim = q@qT/sqrt(QK); top-16; H row (one 64-thread block per (b,n)) ----
__global__ void __launch_bounds__(64)
topk_H_kernel(const float* __restrict__ q, float* __restrict__ H) {
  const int row = blockIdx.x;        // b*64 + n
  const int b = row >> 6, n = row & 63;
  const int m = threadIdx.x;         // 0..63
  __shared__ float sim[64];
  __shared__ int sel[64];
  const float* qb = q + (size_t)b * 64 * 64;
  float dot = 0.f;
  for (int k = 0; k < 64; ++k) dot += qb[n * 64 + k] * qb[m * 64 + k];
  sim[m] = dot * 0.125f;             // 1/sqrt(64)
  sel[m] = 0;
  __syncthreads();
  if (m == 0) {
    for (int it = 0; it < 16; ++it) {
      int best = 0; float bv = -__builtin_inff();
      for (int j = 0; j < 64; ++j)
        if (!sel[j] && sim[j] > bv) { bv = sim[j]; best = j; }
      sel[best] = 1;                 // ties -> lowest index (matches top_k)
    }
  }
  __syncthreads();
  H[(size_t)row * 64 + m] = sel[m] ? 1.0f : 0.0f;
}

// ---- Hw = softmax(aw*H, axis=2) * H, in place over aw ----------------------
__global__ void __launch_bounds__(64)
softmax_mask_kernel(const float* __restrict__ H, float* __restrict__ aw) {
  const int row = blockIdx.x;        // b*64 + e
  const int m = threadIdx.x;
  __shared__ float sv[64];
  __shared__ float maxv, sumv;
  const float h = H[(size_t)row * 64 + m];
  const float x = aw[(size_t)row * 64 + m] * h;   // zeros included, like ref
  sv[m] = x;
  __syncthreads();
  if (m == 0) {
    float mx = sv[0];
    for (int j = 1; j < 64; ++j) mx = fmaxf(mx, sv[j]);
    maxv = mx;
  }
  __syncthreads();
  const float ex = __expf(x - maxv);
  sv[m] = ex;
  __syncthreads();
  if (m == 0) {
    float sm = 0.f;
    for (int j = 0; j < 64; ++j) sm += sv[j];
    sumv = sm;
  }
  __syncthreads();
  aw[(size_t)row * 64 + m] = (ex / sumv) * h;
}

// ---- nf[:, 256:512] = af / 64 ---------------------------------------------
__global__ void __launch_bounds__(256)
pack_nf_kernel(const float* __restrict__ af, float* __restrict__ nf) {
  const int idx = (int)(blockIdx.x * blockDim.x + threadIdx.x);  // 4096*256
  const int row = idx >> 8, col = idx & 255;
  nf[(size_t)row * 512 + 256 + col] = af[idx] * (1.0f / 64.0f);
}

// ---------------------------------------------------------------------------
extern "C" void kernel_launch(void* const* d_in, const int* in_sizes, int n_in,
                              void* d_out, int out_size, void* d_ws, size_t ws_size,
                              hipStream_t stream) {
  (void)in_sizes; (void)n_in; (void)out_size; (void)ws_size;
  const float* x   = (const float*)d_in[0];
  const float* We  = (const float*)d_in[1];
  const float* be  = (const float*)d_in[2];
  const float* Wq  = (const float*)d_in[3];
  const float* bq  = (const float*)d_in[4];
  const float* Wn1 = (const float*)d_in[5];
  const float* bn1 = (const float*)d_in[6];
  const float* Wn2 = (const float*)d_in[7];
  const float* bn2 = (const float*)d_in[8];
  const float* Wa1 = (const float*)d_in[9];
  const float* ba1 = (const float*)d_in[10];
  const float* Wa2 = (const float*)d_in[11];
  const float* ba2 = (const float*)d_in[12];
  const float* Wd1 = (const float*)d_in[13];
  const float* bd1 = (const float*)d_in[14];
  const float* Wd2 = (const float*)d_in[15];
  const float* bd2 = (const float*)d_in[16];

  float* out = (float*)d_out;                       // [4096,256]
  float* Hmat = out + (size_t)4096 * 256;           // H output slot [B,64,64]
  float* ws = (float*)d_ws;

  // workspace layout (floats), with liveness-based reuse
  float* AF  = ws;                                  // [4096,256]
  float* Q   = ws + 1048576;                        // [4096,64]   dead after topk
  float* H1  = ws + 1310720;                        // [4096,512]  dead after xm
  float* XM  = ws + 3407872;                        // [4096,256]
  float* EI  = ws + 1048576;                        // [4096,256]  reuses Q+H1 head
  float* AW  = ws + 2097152;                        // [64,64,64]  reuses H1 mid
  float* EDG = ws + 2359296;                        // [4096,256]  reuses H1 tail
  float* NF  = ws + 4456448;                        // [4096,512]
  float* HD1 = ws;                                  // [4096,512]  reuses AF+EI

  // packed bf16 weights (WMMA B-fragment order), after NF: ~1.4 MB
  bf16_t* PK   = (bf16_t*)(ws + 6553600);
  bf16_t* pWe  = PK;                                // 128x256 -> 32768
  bf16_t* pWq  = PK + 32768;                        // 128x64  -> 8192
  bf16_t* pWn1 = PK + 40960;                        // 256x512 -> 131072
  bf16_t* pWn2 = PK + 172032;                       // 512x256 -> 131072
  bf16_t* pWa1 = PK + 303104;                       // 512x32  -> 16384
  bf16_t* pWd1 = PK + 319488;                       // 512x512 -> 262144
  bf16_t* pWd2 = PK + 581632;                       // 512x256 -> 131072

  auto packw = [&](const float* W, bf16_t* dst, int K, int N) {
    const int total = K * N;
    hipLaunchKernelGGL(pack_w_kernel, dim3((total + 255) / 256), dim3(256), 0,
                       stream, W, dst, N, N / 16, total);
  };
  packw(We,  pWe,  128, 256);
  packw(Wq,  pWq,  128, 64);
  packw(Wn1, pWn1, 256, 512);
  packw(Wn2, pWn2, 512, 256);
  packw(Wa1, pWa1, 512, 32);
  packw(Wd1, pWd1, 512, 512);
  packw(Wd2, pWd2, 512, 256);

  // packed-weight GEMM (TRANSA=0, PACKEDB=1, NBLK=4)
  auto gemmW = [&](const float* A, const bf16_t* pB, const float* bias, float* D,
                   int M, int N, int K, int lda, int ldd, int relu) {
    const int waves = (M / 16) * (N / 64);
    hipLaunchKernelGGL((gemm_bf16_wmma<0, 1, 4>), dim3((waves + 7) / 8), dim3(256),
                       0, stream, A, (const float*)nullptr, pB, bias, D,
                       M, N, K, lda, 0, ldd, 0LL, 0LL, 0LL, 1, relu, 1.0f);
  };
  // f32-B batched GEMM (NBLK=4)
  auto gemmB = [&](const float* A, const float* Bw, float* D,
                   int M, int N, int K, int lda, int ldb, int ldd,
                   long long sA, long long sB, long long sD, int batches,
                   int transA, float scale) {
    const int waves = (M / 16) * (N / 64) * batches;
    if (transA)
      hipLaunchKernelGGL((gemm_bf16_wmma<1, 0, 4>), dim3((waves + 7) / 8),
                         dim3(256), 0, stream, A, Bw, (const bf16_t*)nullptr,
                         (const float*)nullptr, D, M, N, K, lda, ldb, ldd,
                         sA, sB, sD, batches, 0, scale);
    else
      hipLaunchKernelGGL((gemm_bf16_wmma<0, 0, 4>), dim3((waves + 7) / 8),
                         dim3(256), 0, stream, A, Bw, (const bf16_t*)nullptr,
                         (const float*)nullptr, D, M, N, K, lda, ldb, ldd,
                         sA, sB, sD, batches, 0, scale);
  };

  // 1) af = relu(x @ We + be)
  gemmW(x, pWe, be, AF, 4096, 256, 128, 128, 256, 1);
  // 2) q = x @ Wq + bq
  gemmW(x, pWq, bq, Q, 4096, 64, 128, 128, 64, 0);
  // 3) sim + top-16 -> H (written directly into output slot)
  hipLaunchKernelGGL(topk_H_kernel, dim3(4096), dim3(64), 0, stream, Q, Hmat);
  // 4) h1 = relu(af @ Wn1 + bn1)
  gemmW(AF, pWn1, bn1, H1, 4096, 512, 256, 256, 512, 1);
  // 5) xm = relu(h1 @ Wn2 + bn2)
  gemmW(H1, pWn2, bn2, XM, 4096, 256, 512, 512, 256, 1);
  // 6) edge_init = H @ xm   (batched over B)
  gemmB(Hmat, XM, EI, 64, 256, 64, 64, 256, 256, 4096, 16384, 16384, 64, 0, 1.0f);
  // 7) aw = attention MLP (fused 512->32->1), never materializes cat
  hipLaunchKernelGGL(attn_mlp_wmma, dim3(2048), dim3(256), 0, stream,
                     XM, EI, pWa1, ba1, Wa2, ba2, AW, 16384);
  // 8) Hw = softmax(aw * H) * H  (in place)
  hipLaunchKernelGGL(softmax_mask_kernel, dim3(4096), dim3(64), 0, stream, Hmat, AW);
  // 9) edges = Hw @ xm  (batched)
  gemmB(AW, XM, EDG, 64, 256, 64, 64, 256, 256, 4096, 16384, 16384, 64, 0, 1.0f);
  // 10) nf[:, :256] = (H^T @ edges) / 64  (batched, transposed A, ldd=512)
  gemmB(Hmat, EDG, NF, 64, 256, 64, 64, 256, 512, 4096, 16384, 32768, 64, 1,
        1.0f / 64.0f);
  // 11) nf[:, 256:] = af / 64
  hipLaunchKernelGGL(pack_nf_kernel, dim3(4096), dim3(256), 0, stream, AF, NF);
  // 12) hd1 = relu(nf @ Wd1 + bd1)
  gemmW(NF, pWd1, bd1, HD1, 4096, 512, 512, 512, 512, 1);
  // 13) out = relu(hd1 @ Wd2 + bd2)
  gemmW(HD1, pWd2, bd2, out, 4096, 256, 512, 512, 256, 1);
}
